// TransformerBlock_71330816852064
// MI455X (gfx1250) — compile-verified, matrix operand order checked by reading, not run on hardware
//
#include <hip/hip_runtime.h>
#include <hip/hip_bf16.h>
#include <math.h>

#define EMBED 1024
#define HEADS 16
#define HD    64
#define FFN   4096
#define BB    2
#define SS    2048
#define MM    (BB * SS)   // 4096 rows

typedef __attribute__((ext_vector_type(16))) __bf16          v16bf;
typedef __attribute__((ext_vector_type(16))) unsigned short  v16u;
typedef __attribute__((ext_vector_type(8)))  unsigned short  v8u;
typedef __attribute__((ext_vector_type(8)))  float           v8f;

#define F_GELU 1
#define F_RES  2

static __device__ __forceinline__ unsigned short f2bf(float f) {
    unsigned int u = __float_as_uint(f);
    u += 0x7FFFu + ((u >> 16) & 1u);     // round-to-nearest-even
    return (unsigned short)(u >> 16);
}

// Flat->LDS offset: for LDS aperture, addr[31:0] is the wave-relative LDS offset.
static __device__ __forceinline__ unsigned lds_off(const void* p) {
    return (unsigned)(unsigned long long)p;
}

// CDNA5 async copy: GLOBAL_LOAD_ASYNC_TO_LDS_B128 (ASYNCcnt-tracked)
static __device__ __forceinline__ void async_ld_b128(unsigned ldsDst, const void* gSrc) {
    asm volatile("global_load_async_to_lds_b128 %0, %1, off"
                 :: "v"(ldsDst), "v"(gSrc) : "memory");
}
static __device__ __forceinline__ void wait_async0() {
    asm volatile("s_wait_asynccnt 0" ::: "memory");
}

// Build a 16x32 bf16 A/B fragment (ISA 7.12.2 layout) from an LDS row:
// lanes 0-15 take K chunks {0..7,16..23}; lanes 16-31 take {8..15,24..31}.
static __device__ __forceinline__ v16bf load_frag(const unsigned short* p, int hi) {
    v8u lo = *(const v8u*)(p + hi);
    v8u hh = *(const v8u*)(p + 16 + hi);
    v16u r;
#pragma unroll
    for (int j = 0; j < 8; ++j) { r[j] = lo[j]; r[8 + j] = hh[j]; }
    return __builtin_bit_cast(v16bf, r);
}

static __device__ __forceinline__ v8f wmma_bf16(v16bf a, v16bf b, v8f c) {
    return __builtin_amdgcn_wmma_f32_16x16x32_bf16(
        false, a, false, b, (short)0, c, false, false);
}

// ---------------------------------------------------------------- LayerNorm
__global__ __launch_bounds__(256)
void ln_bf16_kernel(const float* __restrict__ x, const float* __restrict__ g,
                    const float* __restrict__ be, unsigned short* __restrict__ out) {
    const int row = blockIdx.x;
    const int tid = threadIdx.x;
    const float* xr = x + (size_t)row * EMBED;
    __shared__ float red[256];
    float loc[4];
    float s = 0.f;
#pragma unroll
    for (int i = 0; i < 4; ++i) { loc[i] = xr[tid + i * 256]; s += loc[i]; }
    red[tid] = s; __syncthreads();
    for (int off = 128; off > 0; off >>= 1) {
        if (tid < off) red[tid] += red[tid + off];
        __syncthreads();
    }
    const float mu = red[0] * (1.0f / EMBED);
    __syncthreads();
    float vs = 0.f;
#pragma unroll
    for (int i = 0; i < 4; ++i) { float d = loc[i] - mu; vs += d * d; }
    red[tid] = vs; __syncthreads();
    for (int off = 128; off > 0; off >>= 1) {
        if (tid < off) red[tid] += red[tid + off];
        __syncthreads();
    }
    const float rs = rsqrtf(red[0] * (1.0f / EMBED) + 1e-5f);
#pragma unroll
    for (int i = 0; i < 4; ++i) {
        const int c = tid + i * 256;
        out[(size_t)row * EMBED + c] = f2bf((loc[i] - mu) * rs * g[c] + be[c]);
    }
}

// -------------------------------------------- Weight convert + transpose (KxN -> NxK bf16)
__global__ __launch_bounds__(256)
void wtrans_kernel(const float* __restrict__ W, unsigned short* __restrict__ Wt,
                   int Kd, int Nd) {
    const int idx = blockIdx.x * 256 + threadIdx.x;
    if (idx >= Kd * Nd) return;
    const int k = idx / Nd, n = idx % Nd;
    Wt[(size_t)n * Kd + k] = f2bf(W[idx]);
}

// ---------------------------------------------------------------- GEMM (bf16 WMMA)
// C[M,N] = A[M,K] * Bt[N,K]^T + bias (+res) (gelu?) -> outf (f32) and/or outb (bf16)
// Double-buffered LDS staging via GLOBAL_LOAD_ASYNC_TO_LDS_B128.
__global__ __launch_bounds__(256)
void gemm_bf16_kernel(const unsigned short* __restrict__ A,
                      const unsigned short* __restrict__ Bt,
                      const float* __restrict__ bias,
                      const float* __restrict__ res,
                      float* __restrict__ outf,
                      unsigned short* __restrict__ outb,
                      int Mdim, int Ndim, int Kdim, int flags) {
    constexpr int LK = 40;                       // 32 + 8 pad (bank-conflict free)
    __shared__ alignas(16) unsigned short As[2][128 * LK];
    __shared__ alignas(16) unsigned short Bs[2][128 * LK];

    const int tid  = threadIdx.x;
    const int lane = tid & 31, w = tid >> 5;
    const int l16  = lane & 15, hi = (lane >> 4) << 3;
    const int wm   = (w & 1) * 64;               // wave row offset in 128-tile
    const int wn   = (w >> 1) * 32;              // wave col offset
    const int n0   = blockIdx.x * 128;
    const int m0   = blockIdx.y * 128;

    // Per-thread staging coordinates (2 chunks each for A and B)
    const int r0 = tid >> 2,        c0 = (tid & 3) << 3;           // idx = tid
    const int r1 = (tid + 256) >> 2, c1 = ((tid + 256) & 3) << 3;  // idx = tid+256
    const unsigned short* gA0 = A  + (size_t)(m0 + r0) * Kdim + c0;
    const unsigned short* gA1 = A  + (size_t)(m0 + r1) * Kdim + c1;
    const unsigned short* gB0 = Bt + (size_t)(n0 + r0) * Kdim + c0;
    const unsigned short* gB1 = Bt + (size_t)(n0 + r1) * Kdim + c1;

    v8f acc[4][2] = {};

    // Prologue: async-stage tile kt=0 into buffer 0
    async_ld_b128(lds_off(&As[0][r0 * LK + c0]), gA0);
    async_ld_b128(lds_off(&As[0][r1 * LK + c1]), gA1);
    async_ld_b128(lds_off(&Bs[0][r0 * LK + c0]), gB0);
    async_ld_b128(lds_off(&Bs[0][r1 * LK + c1]), gB1);
    wait_async0();
    __syncthreads();

    int buf = 0;
    for (int kt = 0; kt < Kdim; kt += 32) {
        // Issue next tile into the other buffer while computing this one
        if (kt + 32 < Kdim) {
            const int kn = kt + 32;
            async_ld_b128(lds_off(&As[buf ^ 1][r0 * LK + c0]), gA0 + kn);
            async_ld_b128(lds_off(&As[buf ^ 1][r1 * LK + c1]), gA1 + kn);
            async_ld_b128(lds_off(&Bs[buf ^ 1][r0 * LK + c0]), gB0 + kn);
            async_ld_b128(lds_off(&Bs[buf ^ 1][r1 * LK + c1]), gB1 + kn);
        }

        v16bf aF[4], bF[2];
#pragma unroll
        for (int mt = 0; mt < 4; ++mt)
            aF[mt] = load_frag(&As[buf][(wm + mt * 16 + l16) * LK], hi);
#pragma unroll
        for (int nt = 0; nt < 2; ++nt)
            bF[nt] = load_frag(&Bs[buf][(wn + nt * 16 + l16) * LK], hi);
#pragma unroll
        for (int mt = 0; mt < 4; ++mt)
#pragma unroll
            for (int nt = 0; nt < 2; ++nt)
                acc[mt][nt] = wmma_bf16(aF[mt], bF[nt], acc[mt][nt]);

        wait_async0();           // our next-tile writes have landed in LDS
        __syncthreads();         // everyone's have; safe to switch buffers
        buf ^= 1;
    }

    // Epilogue: C/D layout row = r + 8*(lane>=16), col = lane&15
#pragma unroll
    for (int mt = 0; mt < 4; ++mt)
#pragma unroll
        for (int nt = 0; nt < 2; ++nt)
#pragma unroll
            for (int r = 0; r < 8; ++r) {
                const int gm = m0 + wm + mt * 16 + r + hi;
                const int gn = n0 + wn + nt * 16 + l16;
                float v = acc[mt][nt][r] + bias[gn];
                if (flags & F_GELU) v = 0.5f * v * (1.0f + erff(v * 0.70710678f));
                if (flags & F_RES)  v += res[(size_t)gm * Ndim + gn];
                if (outf) outf[(size_t)gm * Ndim + gn] = v;
                if (outb) outb[(size_t)gm * Ndim + gn] = f2bf(v);
            }
}

// ---------------------------------------------------------------- RoPE (f32 -> bf16)
__global__ __launch_bounds__(256)
void rope_kernel(const float* __restrict__ qf, const float* __restrict__ kf,
                 const float* __restrict__ cosb, const float* __restrict__ sinb,
                 unsigned short* __restrict__ qb, unsigned short* __restrict__ kb) {
    const size_t idx = (size_t)blockIdx.x * 256 + threadIdx.x;
    if (idx >= (size_t)MM * EMBED) return;
    const int d = (int)(idx & 63);
    const int s = (int)((idx >> 10) & (SS - 1));
    const float c  = cosb[s * HD + d];
    const float sn = sinb[s * HD + d];
    const size_t base = idx - d;
    const float qv = qf[idx], kv = kf[idx];
    const float qr = (d < 32) ? -qf[base + d + 32] : qf[base + d - 32];
    const float kr = (d < 32) ? -kf[base + d + 32] : kf[base + d - 32];
    qb[idx] = f2bf(qv * c + qr * sn);
    kb[idx] = f2bf(kv * c + kr * sn);
}

// ---------------------------------------------------------------- Flash attention
// Q/K/V/O laid out [b][s][h*64+d] bf16. Block = 128 q rows, 8 waves x 16 rows.
__global__ __launch_bounds__(256)
void attn_kernel(const unsigned short* __restrict__ Q,
                 const unsigned short* __restrict__ K,
                 const unsigned short* __restrict__ V,
                 unsigned short* __restrict__ O) {
    constexpr int LD = 72;                       // 64 + 8 pad
    __shared__ alignas(16) unsigned short Qs[128 * LD];
    __shared__ alignas(16) unsigned short Ks[64 * LD];
    __shared__ alignas(16) unsigned short Vt[64 * LD];     // transposed: [d][s]
    __shared__ alignas(16) unsigned short Ps[8 * 16 * LD]; // per-wave P staging

    const int tid  = threadIdx.x;
    const int lane = tid & 31, w = tid >> 5;
    const int l16  = lane & 15, hi = (lane >> 4) << 3;
    const int qb0  = blockIdx.x * 128;
    const int h    = blockIdx.y;
    const int b    = blockIdx.z;
    const size_t rowBase = (size_t)b * SS;
    const int hoff = h * HD;

    // Load Q tile 128x64 (async -> LDS)
#pragma unroll
    for (int it = 0; it < 4; ++it) {
        const int idx = tid + it * 256;
        const int row = idx >> 3, c8 = (idx & 7) << 3;
        async_ld_b128(lds_off(&Qs[row * LD + c8]),
                      Q + (rowBase + qb0 + row) * EMBED + hoff + c8);
    }
    wait_async0();
    __syncthreads();

    v16bf aQ[2];
    aQ[0] = load_frag(&Qs[(w * 16 + l16) * LD + 0],  hi);
    aQ[1] = load_frag(&Qs[(w * 16 + l16) * LD + 32], hi);

    v8f of[4] = {};
    float m_run[8], l_run[8];
#pragma unroll
    for (int r = 0; r < 8; ++r) { m_run[r] = -3e38f; l_run[r] = 0.f; }

    const int nkb = qb0 / 64 + 2;                // causal: key blocks covering keys <= q
    for (int jb = 0; jb < nkb; ++jb) {
        const int k0 = jb * 64;
        __syncthreads();                          // protect Ks/Vt reuse
#pragma unroll
        for (int it = 0; it < 2; ++it) {
            const int idx = tid + it * 256;
            const int row = idx >> 3, c8 = (idx & 7) << 3;
            async_ld_b128(lds_off(&Ks[row * LD + c8]),
                          K + (rowBase + k0 + row) * EMBED + hoff + c8);
            v8u vd = *(const v8u*)(V + (rowBase + k0 + row) * EMBED + hoff + c8);
#pragma unroll
            for (int j = 0; j < 8; ++j) Vt[(c8 + j) * LD + row] = vd[j];
        }
        wait_async0();
        __syncthreads();

        // scores: Q (16x64) x K^T (64x64) -> 16x64
        v8f sc[4] = {};
#pragma unroll
        for (int ks = 0; ks < 2; ++ks)
#pragma unroll
            for (int nt = 0; nt < 4; ++nt) {
                v16bf bK = load_frag(&Ks[(nt * 16 + l16) * LD + ks * 32], hi);
                sc[nt] = wmma_bf16(aQ[ks], bK, sc[nt]);
            }

        // scale + causal mask
#pragma unroll
        for (int nt = 0; nt < 4; ++nt)
#pragma unroll
            for (int r = 0; r < 8; ++r) {
                float s = sc[nt][r] * 0.125f;     // 1/sqrt(64)
                const int qg = qb0 + w * 16 + r + hi;
                const int kg = k0 + nt * 16 + l16;
                sc[nt][r] = (kg > qg) ? -3e38f : s;
            }

        // row max across the 16 lanes of each half-group
        float rmax[8];
#pragma unroll
        for (int r = 0; r < 8; ++r) {
            float mx = sc[0][r];
#pragma unroll
            for (int nt = 1; nt < 4; ++nt) mx = fmaxf(mx, sc[nt][r]);
            rmax[r] = mx;
        }
#pragma unroll
        for (int msk = 1; msk < 16; msk <<= 1)
#pragma unroll
            for (int r = 0; r < 8; ++r)
                rmax[r] = fmaxf(rmax[r], __shfl_xor(rmax[r], msk, 32));

        float alpha[8], rsum[8];
#pragma unroll
        for (int r = 0; r < 8; ++r) {
            const float nm = fmaxf(fmaxf(m_run[r], rmax[r]), -1e30f);
            alpha[r] = __expf(m_run[r] - nm);
            m_run[r] = nm;
            rsum[r] = 0.f;
        }
        // exp -> P (staged per wave in LDS as bf16, A-fragment consumable)
#pragma unroll
        for (int nt = 0; nt < 4; ++nt)
#pragma unroll
            for (int r = 0; r < 8; ++r) {
                const float p = __expf(sc[nt][r] - m_run[r]);
                rsum[r] += p;
                Ps[(w * 16 + r + hi) * LD + nt * 16 + l16] = f2bf(p);
            }
#pragma unroll
        for (int msk = 1; msk < 16; msk <<= 1)
#pragma unroll
            for (int r = 0; r < 8; ++r)
                rsum[r] += __shfl_xor(rsum[r], msk, 32);
#pragma unroll
        for (int r = 0; r < 8; ++r) l_run[r] = l_run[r] * alpha[r] + rsum[r];
        // rescale O accumulator (same row layout as scores)
#pragma unroll
        for (int nt = 0; nt < 4; ++nt)
#pragma unroll
            for (int r = 0; r < 8; ++r) of[nt][r] *= alpha[r];
        __syncthreads();                          // make P visible (ds wait)

        // O += P (16x64) x V (64x64)
#pragma unroll
        for (int ks = 0; ks < 2; ++ks) {
            v16bf aP = load_frag(&Ps[(w * 16 + l16) * LD + ks * 32], hi);
#pragma unroll
            for (int nt = 0; nt < 4; ++nt) {
                v16bf bV = load_frag(&Vt[(nt * 16 + l16) * LD + ks * 32], hi);
                of[nt] = wmma_bf16(aP, bV, of[nt]);
            }
        }
    }

    // finalize: divide by l, write bf16
#pragma unroll
    for (int r = 0; r < 8; ++r) {
        const float inv = 1.0f / l_run[r];
        const int qg = qb0 + w * 16 + r + hi;
#pragma unroll
        for (int nt = 0; nt < 4; ++nt)
            O[(rowBase + qg) * EMBED + hoff + nt * 16 + l16] = f2bf(of[nt][r] * inv);
    }
}

// ============================================================================
extern "C" void kernel_launch(void* const* d_in, const int* in_sizes, int n_in,
                              void* d_out, int out_size, void* d_ws, size_t ws_size,
                              hipStream_t stream) {
    const float* x        = (const float*)d_in[0];
    const float* rope_cos = (const float*)d_in[1];
    const float* rope_sin = (const float*)d_in[2];
    const float* Wq = (const float*)d_in[3];  const float* bq = (const float*)d_in[4];
    const float* Wk = (const float*)d_in[5];  const float* bk = (const float*)d_in[6];
    const float* Wv = (const float*)d_in[7];  const float* bv = (const float*)d_in[8];
    const float* Wo = (const float*)d_in[9];  const float* bo = (const float*)d_in[10];
    const float* W1 = (const float*)d_in[11]; const float* b1 = (const float*)d_in[12];
    const float* W2 = (const float*)d_in[13]; const float* b2 = (const float*)d_in[14];
    const float* g1 = (const float*)d_in[15]; const float* be1 = (const float*)d_in[16];
    const float* g2 = (const float*)d_in[17]; const float* be2 = (const float*)d_in[18];
    float* out = (float*)d_out;
    char* ws = (char*)d_ws;

    size_t off = 0;
    auto nxt = [&](size_t bytes) {
        size_t o = off; off += (bytes + 255) & ~(size_t)255; return o;
    };
    unsigned short* wqT = (unsigned short*)(ws + nxt((size_t)EMBED * EMBED * 2));
    unsigned short* wkT = (unsigned short*)(ws + nxt((size_t)EMBED * EMBED * 2));
    unsigned short* wvT = (unsigned short*)(ws + nxt((size_t)EMBED * EMBED * 2));
    unsigned short* woT = (unsigned short*)(ws + nxt((size_t)EMBED * EMBED * 2));
    unsigned short* w1T = (unsigned short*)(ws + nxt((size_t)EMBED * FFN * 2));
    unsigned short* w2T = (unsigned short*)(ws + nxt((size_t)FFN * EMBED * 2));
    unsigned short* n1  = (unsigned short*)(ws + nxt((size_t)MM * EMBED * 2));
    float*          qf  = (float*)         (ws + nxt((size_t)MM * EMBED * 4));
    float*          kf  = (float*)         (ws + nxt((size_t)MM * EMBED * 4));
    unsigned short* qb_ = (unsigned short*)(ws + nxt((size_t)MM * EMBED * 2));
    unsigned short* kb_ = (unsigned short*)(ws + nxt((size_t)MM * EMBED * 2));
    unsigned short* vb_ = (unsigned short*)(ws + nxt((size_t)MM * EMBED * 2));
    unsigned short* ab  = (unsigned short*)(ws + nxt((size_t)MM * EMBED * 2));
    float*          x1  = (float*)         (ws + nxt((size_t)MM * EMBED * 4));
    unsigned short* n2  = (unsigned short*)(ws + nxt((size_t)MM * EMBED * 2));
    unsigned short* hb  = (unsigned short*)(ws + nxt((size_t)MM * FFN * 2));
    (void)ws_size; (void)in_sizes; (void)n_in; (void)out_size;

    // Weight bf16 transpose (NxK layout so GEMM A/B tiles load identically)
    wtrans_kernel<<<(EMBED * EMBED + 255) / 256, 256, 0, stream>>>(Wq, wqT, EMBED, EMBED);
    wtrans_kernel<<<(EMBED * EMBED + 255) / 256, 256, 0, stream>>>(Wk, wkT, EMBED, EMBED);
    wtrans_kernel<<<(EMBED * EMBED + 255) / 256, 256, 0, stream>>>(Wv, wvT, EMBED, EMBED);
    wtrans_kernel<<<(EMBED * EMBED + 255) / 256, 256, 0, stream>>>(Wo, woT, EMBED, EMBED);
    wtrans_kernel<<<(EMBED * FFN   + 255) / 256, 256, 0, stream>>>(W1, w1T, EMBED, FFN);
    wtrans_kernel<<<(FFN * EMBED   + 255) / 256, 256, 0, stream>>>(W2, w2T, FFN, EMBED);

    // LN1
    ln_bf16_kernel<<<MM, 256, 0, stream>>>(x, g1, be1, n1);

    // QKV projections
    dim3 gEE(EMBED / 128, MM / 128);
    gemm_bf16_kernel<<<gEE, 256, 0, stream>>>(n1, wqT, bq, nullptr, qf, nullptr, MM, EMBED, EMBED, 0);
    gemm_bf16_kernel<<<gEE, 256, 0, stream>>>(n1, wkT, bk, nullptr, kf, nullptr, MM, EMBED, EMBED, 0);
    gemm_bf16_kernel<<<gEE, 256, 0, stream>>>(n1, wvT, bv, nullptr, nullptr, vb_, MM, EMBED, EMBED, 0);

    // RoPE on Q,K (f32 -> bf16)
    rope_kernel<<<((size_t)MM * EMBED + 255) / 256, 256, 0, stream>>>(
        qf, kf, rope_cos, rope_sin, qb_, kb_);

    // Causal flash attention
    attn_kernel<<<dim3(SS / 128, HEADS, BB), 256, 0, stream>>>(qb_, kb_, vb_, ab);

    // O projection + residual
    gemm_bf16_kernel<<<gEE, 256, 0, stream>>>(ab, woT, bo, x, x1, nullptr, MM, EMBED, EMBED, F_RES);

    // LN2
    ln_bf16_kernel<<<MM, 256, 0, stream>>>(x1, g2, be2, n2);

    // FFN
    dim3 gEF(FFN / 128, MM / 128);
    gemm_bf16_kernel<<<gEF, 256, 0, stream>>>(n2, w1T, b1, nullptr, nullptr, hb, MM, FFN, EMBED, F_GELU);
    gemm_bf16_kernel<<<gEE, 256, 0, stream>>>(hb, w2T, b2, x1, out, nullptr, MM, EMBED, FFN, F_RES);
}
